// AlbertMultiHeadAttention_2723009266443
// MI455X (gfx1250) — matrix operand order repaired and checked by reading, not compiled, not run on hardware
//
#include <hip/hip_runtime.h>
#include <hip/hip_bf16.h>
#include <math.h>

#define B_ 8
#define S_ 1024
#define D_ 768
#define H_ 12
#define DK_ 64
#define EPS_ 1e-8f

typedef __attribute__((ext_vector_type(16))) __bf16 v16bf;
typedef __attribute__((ext_vector_type(8)))  float  v8f;
typedef __attribute__((ext_vector_type(4)))  float  fx4;
typedef __attribute__((ext_vector_type(4)))  int    i32x4;

union FragAB {
  v16bf v;
  unsigned short u[16];
  uint4 q[2];
};

__device__ __forceinline__ unsigned short f2bf(float f) {
  unsigned u = __float_as_uint(f);
  u += 0x7FFFu + ((u >> 16) & 1u);   // round-to-nearest-even
  return (unsigned short)(u >> 16);
}

// ---- async global->LDS staging (CDNA5 ASYNCcnt path), guarded fallback ----
#if defined(__HIP_DEVICE_COMPILE__) && __has_builtin(__builtin_amdgcn_global_load_async_to_lds_b128)
#define HAVE_ASYNC_LDS 1
#endif

__device__ __forceinline__ void async_copy16(const void* g, void* l) {
#ifdef HAVE_ASYNC_LDS
  __builtin_amdgcn_global_load_async_to_lds_b128(
      (__attribute__((address_space(1))) i32x4*)g,
      (__attribute__((address_space(3))) i32x4*)l, 0, 0);
#else
  *(uint4*)l = *(const uint4*)g;
#endif
}
__device__ __forceinline__ void async_join() {
#ifdef HAVE_ASYNC_LDS
#if __has_builtin(__builtin_amdgcn_s_wait_asynccnt)
  __builtin_amdgcn_s_wait_asynccnt(0);
#else
  asm volatile("s_wait_asynccnt 0x0" ::: "memory");
#endif
#endif
}

// ---------------- fp32 -> bf16 elementwise ----------------
__global__ void cvt_bf16_kernel(const float* __restrict__ in,
                                unsigned short* __restrict__ out, int n) {
  int i = blockIdx.x * blockDim.x + threadIdx.x;
  int stride = gridDim.x * blockDim.x;
  for (; i < n; i += stride) out[i] = f2bf(in[i]);
}

// ---------------- weight convert + transpose: WT[n][k] = bf16(W[k][n]) ----------------
__global__ __launch_bounds__(256) void wtrans_kernel(const float* __restrict__ W,
                                                     unsigned short* __restrict__ WT) {
  __shared__ float tile[32][33];
  int tc = blockIdx.x;   // n tile
  int tr = blockIdx.y;   // k tile
  int t = threadIdx.x;
  int r = t >> 3, c4 = (t & 7) * 4;
  const float* src = W + (size_t)(tr * 32 + r) * D_ + tc * 32 + c4;
  float4 v = *(const float4*)src;
  tile[r][c4 + 0] = v.x; tile[r][c4 + 1] = v.y;
  tile[r][c4 + 2] = v.z; tile[r][c4 + 3] = v.w;
  __syncthreads();
  unsigned short* dst = WT + (size_t)(tc * 32 + r) * D_ + tr * 32 + c4;
  ushort4 o;
  o.x = f2bf(tile[c4 + 0][r]); o.y = f2bf(tile[c4 + 1][r]);
  o.z = f2bf(tile[c4 + 2][r]); o.w = f2bf(tile[c4 + 3][r]);
  *(ushort4*)dst = o;
}

// ---------------- projection GEMM: Y = Xb @ W + bias, head-split output ----------------
// mode 0/1: out[b,h,s,dk] (Q,K)   mode 2: out[b,h,dk,s] (V transposed)
__global__ __launch_bounds__(256) void proj_gemm_kernel(
    const unsigned short* __restrict__ Xb,   // [8192,768] bf16
    const unsigned short* __restrict__ WT,   // [768,768] bf16 (W transposed)
    const float* __restrict__ bias,          // [768]
    unsigned short* __restrict__ out, int mode) {
  __shared__ unsigned short ldsA[64 * 32];
  __shared__ unsigned short ldsB[128 * 32];
  const int t = threadIdx.x;
  const int wave = t >> 5, lane = t & 31;
  const int r = lane & 15, kh = lane >> 4;
  const int wm = wave >> 2, wn = wave & 3;    // 2 x 4 wave grid
  const int bm = blockIdx.x * 64;
  const int bn = blockIdx.y * 128;

  v8f acc[2][2] = {};
  for (int kk = 0; kk < D_; kk += 32) {
    {   // stage A 64x32 via async DMA to LDS
      int row = t >> 2, cg = (t & 3) * 8;
      async_copy16(Xb + (size_t)(bm + row) * D_ + kk + cg,
                   ldsA + row * 32 + cg);
    }
    {   // stage B 128x32 (row n of WT, contiguous k)
      int row = t >> 1, hg = (t & 1) * 16;
      const unsigned short* src = WT + (size_t)(bn + row) * D_ + kk + hg;
      unsigned short* dst = ldsB + row * 32 + hg;
      async_copy16(src, dst);
      async_copy16(src + 8, dst + 8);
    }
    async_join();
    __syncthreads();
    FragAB a[2], b[2];
#pragma unroll
    for (int mi = 0; mi < 2; ++mi) {
      const unsigned short* p = ldsA + (wm * 32 + mi * 16 + r) * 32;
      a[mi].q[0] = *(const uint4*)(p + kh * 16);
      a[mi].q[1] = *(const uint4*)(p + kh * 16 + 8);
    }
#pragma unroll
    for (int ni = 0; ni < 2; ++ni) {
      const unsigned short* p = ldsB + (wn * 32 + ni * 16 + r) * 32 + kh * 16;
      b[ni].q[0] = ((const uint4*)p)[0];
      b[ni].q[1] = ((const uint4*)p)[1];
    }
#pragma unroll
    for (int mi = 0; mi < 2; ++mi)
#pragma unroll
      for (int ni = 0; ni < 2; ++ni)
        acc[mi][ni] = __builtin_amdgcn_wmma_f32_16x16x32_bf16(
            false, a[mi].v, false, b[ni].v, (short)0, acc[mi][ni], false, false);
    __syncthreads();
  }
#pragma unroll
  for (int mi = 0; mi < 2; ++mi)
#pragma unroll
    for (int ni = 0; ni < 2; ++ni)
#pragma unroll
      for (int j = 0; j < 8; ++j) {
        int m = bm + wm * 32 + mi * 16 + j + 8 * kh;
        int n = bn + wn * 32 + ni * 16 + r;
        float val = acc[mi][ni][j] + bias[n];
        int bb = m >> 10, s = m & 1023;
        int h = n >> 6, dk = n & 63;
        size_t idx;
        if (mode == 2) idx = ((size_t)(bb * H_ + h) * DK_ + dk) * S_ + s;
        else           idx = ((size_t)(bb * H_ + h) * S_ + s) * DK_ + dk;
        out[idx] = f2bf(val);
      }
}

// ---------------- fused attention: scores -> softmax -> attn out + ctx ----------------
__global__ __launch_bounds__(256) void attn_kernel(
    const unsigned short* __restrict__ Qh,  // [B,H,S,DK] bf16
    const unsigned short* __restrict__ Kh,  // [B,H,S,DK] bf16
    const unsigned short* __restrict__ Vt,  // [B,H,DK,S] bf16
    const float* __restrict__ mask,         // [B,1,1,S]
    float* __restrict__ attn_out,           // [B,H,S,S]
    unsigned short* __restrict__ ctx) {     // [B,S,D] bf16
  extern __shared__ float sc[];             // [32][1024] = 128 KB
  __shared__ float red[256];
  __shared__ float rowmax[32], rowsum[32];
  int blk = blockIdx.x;
  int qb = blk & 31;                        // query block of 32 rows
  int bh = blk >> 5;
  int b = bh / H_, h = bh % H_;
  const int t = threadIdx.x, wave = t >> 5, lane = t & 31;
  const int r = lane & 15, kh = lane >> 4;

  const unsigned short* Qbase = Qh + (size_t)bh * S_ * DK_;
  const unsigned short* Kbase = Kh + (size_t)bh * S_ * DK_;
  const unsigned short* Vbase = Vt + (size_t)bh * DK_ * S_;

  // ---- score phase: 32 rows x 1024 cols, wave grid 2 (rows) x 4 (col chunks) ----
  {
    int rh = wave >> 2, chunk = wave & 3;
    int qrow = qb * 32 + rh * 16 + r;
    const unsigned short* qp = Qbase + (size_t)qrow * DK_;
    FragAB qf[2];
#pragma unroll
    for (int s2 = 0; s2 < 2; ++s2) {
      qf[s2].q[0] = *(const uint4*)(qp + s2 * 32 + kh * 16);
      qf[s2].q[1] = *(const uint4*)(qp + s2 * 32 + kh * 16 + 8);
    }
    for (int ct = 0; ct < 16; ++ct) {
      int j0 = chunk * 256 + ct * 16;
      const unsigned short* kp = Kbase + (size_t)(j0 + r) * DK_;
      if (ct + 1 < 16)                      // prefetch next K tile row
        __builtin_prefetch((const void*)(kp + 16 * DK_), 0, 1);
      v8f acc = {};
#pragma unroll
      for (int s2 = 0; s2 < 2; ++s2) {
        FragAB kf;
        kf.q[0] = ((const uint4*)(kp + s2 * 32 + kh * 16))[0];
        kf.q[1] = ((const uint4*)(kp + s2 * 32 + kh * 16))[1];
        acc = __builtin_amdgcn_wmma_f32_16x16x32_bf16(
            false, qf[s2].v, false, kf.v, (short)0, acc, false, false);
      }
      float mterm = (1.0f - mask[b * S_ + j0 + r]) * -10000.0f;
#pragma unroll
      for (int j = 0; j < 8; ++j) {
        int lr = rh * 16 + j + 8 * kh;
        sc[lr * S_ + j0 + r] = acc[j] * 0.125f + mterm;
      }
    }
  }
  __syncthreads();

  // ---- softmax: 8 threads per row ----
  {
    int row = t >> 3, sub = t & 7;
    float* srow = sc + row * S_;
    float m = -3.4e38f;
    for (int i = sub * 128; i < sub * 128 + 128; ++i) m = fmaxf(m, srow[i]);
    red[t] = m;
    __syncthreads();
    if (sub == 0) {
      float mm = red[t];
      for (int k = 1; k < 8; ++k) mm = fmaxf(mm, red[t + k]);
      rowmax[row] = mm;
    }
    __syncthreads();
    float mx = rowmax[row];
    float s = 0.f;
    for (int i = sub * 128; i < sub * 128 + 128; ++i) {
      float e = __expf(srow[i] - mx);
      srow[i] = e; s += e;
    }
    red[t] = s;
    __syncthreads();
    if (sub == 0) {
      float ss = red[t];
      for (int k = 1; k < 8; ++k) ss += red[t + k];
      rowsum[row] = ss;
    }
    __syncthreads();
    float inv = 1.0f / rowsum[row];
    float* outp = attn_out + ((size_t)bh * S_ + qb * 32 + row) * S_;
    for (int i = sub * 128; i < sub * 128 + 128; i += 4) {
      fx4 v;
      v.x = srow[i] * inv; v.y = srow[i + 1] * inv;
      v.z = srow[i + 2] * inv; v.w = srow[i + 3] * inv;
      srow[i] = v.x; srow[i + 1] = v.y; srow[i + 2] = v.z; srow[i + 3] = v.w;
      __builtin_nontemporal_store(v, (fx4*)(outp + i));   // stream 403 MB once
    }
  }
  __syncthreads();

  // ---- ctx = attn @ V: 8 waves -> 2x4 tiles of 16x16 over 32x64 output ----
  {
    int mi = wave >> 2, ni = wave & 3;
    v8f acc = {};
    const unsigned short* vp = Vbase + (size_t)(ni * 16 + r) * S_;
    const float* ap0 = sc + (mi * 16 + r) * S_;
    for (int kk = 0; kk < S_; kk += 32) {
      FragAB a, bfr;
      const float* ap = ap0 + kk;
#pragma unroll
      for (int i = 0; i < 8; ++i) {
        a.u[i]     = f2bf(ap[kh * 16 + i]);
        a.u[8 + i] = f2bf(ap[kh * 16 + 8 + i]);
      }
      bfr.q[0] = ((const uint4*)(vp + kk + kh * 16))[0];
      bfr.q[1] = ((const uint4*)(vp + kk + kh * 16))[1];
      acc = __builtin_amdgcn_wmma_f32_16x16x32_bf16(
          false, a.v, false, bfr.v, (short)0, acc, false, false);
    }
#pragma unroll
    for (int j = 0; j < 8; ++j) {
      int srow = qb * 32 + mi * 16 + j + 8 * kh;
      int d = h * DK_ + ni * 16 + r;
      ctx[(size_t)(b * S_ + srow) * D_ + d] = f2bf(acc[j]);
    }
  }
}

// ---------------- out projection + bias + residual + LayerNorm ----------------
__global__ __launch_bounds__(256) void outproj_ln_kernel(
    const unsigned short* __restrict__ Ctx,  // [8192,768] bf16
    const unsigned short* __restrict__ WoT,  // [768,768] bf16 (Wo transposed)
    const float* __restrict__ bo,
    const float* __restrict__ origin,        // query fp32
    const float* __restrict__ gamma,
    const float* __restrict__ beta,
    float* __restrict__ out) {
  __shared__ unsigned short ldsA[16 * 32];
  __shared__ float ldsO[16 * D_];
  __shared__ float red[256];
  const int t = threadIdx.x, wave = t >> 5, lane = t & 31;
  const int r = lane & 15, kh = lane >> 4;
  const int bm = blockIdx.x * 16;

  v8f acc[6] = {};
  for (int kk = 0; kk < D_; kk += 32) {
    if (t < 64) {   // stage A 16x32: 512 bf16 = 64 x 16B async copies
      int row = t >> 2, col = (t & 3) * 8;
      async_copy16(Ctx + (size_t)(bm + row) * D_ + kk + col,
                   ldsA + row * 32 + col);
    }
    async_join();
    __syncthreads();
    FragAB a;
    const unsigned short* ap = ldsA + r * 32;
    a.q[0] = *(const uint4*)(ap + kh * 16);
    a.q[1] = *(const uint4*)(ap + kh * 16 + 8);
#pragma unroll
    for (int t6 = 0; t6 < 6; ++t6) {
      int n = wave * 96 + t6 * 16 + r;
      const unsigned short* wp = WoT + (size_t)n * D_ + kk + kh * 16;
      FragAB bfr;
      bfr.q[0] = ((const uint4*)wp)[0];
      bfr.q[1] = ((const uint4*)wp)[1];
      acc[t6] = __builtin_amdgcn_wmma_f32_16x16x32_bf16(
          false, a.v, false, bfr.v, (short)0, acc[t6], false, false);
    }
    __syncthreads();
  }
#pragma unroll
  for (int t6 = 0; t6 < 6; ++t6) {
    int n = wave * 96 + t6 * 16 + r;
    float bb = bo[n];
#pragma unroll
    for (int j = 0; j < 8; ++j)
      ldsO[(j + 8 * kh) * D_ + n] = acc[t6][j] + bb;
  }
  __syncthreads();
  {   // residual + LayerNorm: 16 threads per row, 48 cols each
    int row = t >> 4, sub = t & 15;
    float* o = ldsO + row * D_;
    const float* org = origin + (size_t)(bm + row) * D_;
    float s = 0.f;
    for (int i = sub * 48; i < sub * 48 + 48; ++i) {
      float x = o[i] + org[i];
      o[i] = x; s += x;
    }
    red[t] = s;
    __syncthreads();
    if (sub == 0) {
      float ss = 0.f;
      for (int k = 0; k < 16; ++k) ss += red[t + k];
      red[t] = ss * (1.0f / D_);
    }
    __syncthreads();
    float mu = red[t & ~15];
    float v = 0.f;
    for (int i = sub * 48; i < sub * 48 + 48; ++i) {
      float d = o[i] - mu; v += d * d;
    }
    __syncthreads();
    red[t] = v;
    __syncthreads();
    if (sub == 0) {
      float vv = 0.f;
      for (int k = 0; k < 16; ++k) vv += red[t + k];
      red[t] = vv * (1.0f / D_);
    }
    __syncthreads();
    float rstd = rsqrtf(red[t & ~15] + EPS_);
    float* op = out + (size_t)(bm + row) * D_;
    for (int i = sub * 48; i < sub * 48 + 48; ++i)
      op[i] = (o[i] - mu) * rstd * gamma[i] + beta[i];
  }
}

extern "C" void kernel_launch(void* const* d_in, const int* in_sizes, int n_in,
                              void* d_out, int out_size, void* d_ws, size_t ws_size,
                              hipStream_t stream) {
  (void)in_sizes; (void)n_in; (void)out_size; (void)ws_size;
  const float* query = (const float*)d_in[0];
  const float* key_  = (const float*)d_in[1];
  const float* value = (const float*)d_in[2];
  const float* mask  = (const float*)d_in[3];
  const float* Wq = (const float*)d_in[4];
  const float* bq = (const float*)d_in[5];
  const float* Wk = (const float*)d_in[6];
  const float* bk = (const float*)d_in[7];
  const float* Wv = (const float*)d_in[8];
  const float* bv = (const float*)d_in[9];
  const float* Wo = (const float*)d_in[10];
  const float* bo = (const float*)d_in[11];
  const float* gamma = (const float*)d_in[12];
  const float* beta  = (const float*)d_in[13];
  float* out  = (float*)d_out;
  float* attn = out + (size_t)B_ * S_ * D_;

  char* ws = (char*)d_ws;
  size_t off = 0;
  auto alloc = [&](size_t bytes) {
    char* p = ws + off; off += (bytes + 255) & ~(size_t)255; return p;
  };
  const size_t nX = (size_t)B_ * S_ * D_;      // 6,291,456
  unsigned short* qb  = (unsigned short*)alloc(nX * 2);
  unsigned short* kb  = (unsigned short*)alloc(nX * 2);
  unsigned short* vb  = (unsigned short*)alloc(nX * 2);
  unsigned short* WqT = (unsigned short*)alloc((size_t)D_ * D_ * 2);
  unsigned short* WkT = (unsigned short*)alloc((size_t)D_ * D_ * 2);
  unsigned short* WvT = (unsigned short*)alloc((size_t)D_ * D_ * 2);
  unsigned short* WoT = (unsigned short*)alloc((size_t)D_ * D_ * 2);
  unsigned short* QhW = (unsigned short*)alloc(nX * 2);
  unsigned short* KhW = (unsigned short*)alloc(nX * 2);
  unsigned short* VtW = (unsigned short*)alloc(nX * 2);
  unsigned short* CtxW = (unsigned short*)alloc(nX * 2);

  // 1. activation fp32 -> bf16
  cvt_bf16_kernel<<<1024, 256, 0, stream>>>(query, qb, (int)nX);
  cvt_bf16_kernel<<<1024, 256, 0, stream>>>(key_,  kb, (int)nX);
  cvt_bf16_kernel<<<1024, 256, 0, stream>>>(value, vb, (int)nX);
  // 2. weight transpose + convert
  dim3 tg(D_ / 32, D_ / 32);
  wtrans_kernel<<<tg, 256, 0, stream>>>(Wq, WqT);
  wtrans_kernel<<<tg, 256, 0, stream>>>(Wk, WkT);
  wtrans_kernel<<<tg, 256, 0, stream>>>(Wv, WvT);
  wtrans_kernel<<<tg, 256, 0, stream>>>(Wo, WoT);
  // 3. Q/K/V projections (V written transposed per head)
  dim3 gp((B_ * S_) / 64, D_ / 128);
  proj_gemm_kernel<<<gp, 256, 0, stream>>>(qb, WqT, bq, QhW, 0);
  proj_gemm_kernel<<<gp, 256, 0, stream>>>(kb, WkT, bk, KhW, 1);
  proj_gemm_kernel<<<gp, 256, 0, stream>>>(vb, WvT, bv, VtW, 2);
  // 4. fused attention (128 KB dynamic LDS for the 32x1024 score slab)
  attn_kernel<<<B_ * H_ * (S_ / 32), 256, 32 * S_ * sizeof(float), stream>>>(
      QhW, KhW, VtW, mask, attn, CtxW);
  // 5. output projection + residual + LayerNorm
  outproj_ln_kernel<<<(B_ * S_) / 16, 256, 0, stream>>>(
      CtxW, WoT, bo, query, gamma, beta, out);
}
